// SparseAttention_46643344835066
// MI455X (gfx1250) — compile-verified, matrix operand order checked by reading, not run on hardware
//
#include <hip/hip_runtime.h>
#include <hip/hip_bf16.h>

// ---------------------------------------------------------------------------
// Problem constants (fixed by the reference)
// ---------------------------------------------------------------------------
#define BATCH 2
#define SEQ   2048
#define CH    1024
#define NHEAD 8
#define HDIM  128          // CH / NHEAD
#define WHALF 128          // WS/2 : keep |i-j| < 128
#define MTOT  (BATCH*SEQ)  // 4096 rows for the projections

typedef __attribute__((ext_vector_type(16))) __bf16 v16bf;
typedef __attribute__((ext_vector_type(8)))  float  v8f;
typedef __attribute__((ext_vector_type(4)))  int    v4i;

union FragB { v16bf v; uint4 q[2]; };   // 32 bytes: one WMMA bf16 A/B operand

// ---------------------------------------------------------------------------
// CDNA5 async global->LDS copy (ASYNCcnt path), with sync fallback
// ---------------------------------------------------------------------------
#if defined(__AMDGCN__) && __has_builtin(__builtin_amdgcn_global_load_async_to_lds_b128)
#define CDNA5_ASYNC 1
#else
#define CDNA5_ASYNC 0
#endif

#if CDNA5_ASYNC
typedef __attribute__((address_space(1))) v4i gv4i;   // global (AS1), 64-bit ptr
typedef __attribute__((address_space(3))) v4i lv4i;   // LDS (AS3), 32-bit ptr
#endif

static __device__ __forceinline__ void cp16(const unsigned short* g, unsigned short* l) {
#if CDNA5_ASYNC
    // AS1 pointer is numerically the generic global address; AS3 pointer is the
    // low 32 bits of the generic LDS address (aperture|offset form per ISA).
    gv4i* gp = (gv4i*)(unsigned long long)g;
    lv4i* lp = (lv4i*)(unsigned)(unsigned long long)l;
    __builtin_amdgcn_global_load_async_to_lds_b128(gp, lp, 0, 0);
#else
    *(uint4*)l = *(const uint4*)g;
#endif
}

static __device__ __forceinline__ void wait_stage() {
#if CDNA5_ASYNC
# if __has_builtin(__builtin_amdgcn_s_wait_asynccnt)
    __builtin_amdgcn_s_wait_asynccnt(0);
# else
    asm volatile("s_wait_asynccnt 0" ::: "memory");
# endif
#endif
}

static __device__ __forceinline__ unsigned short f2bf(float f) {
    union { float f; unsigned u; } x; x.f = f;
    unsigned r = x.u + 0x7FFFu + ((x.u >> 16) & 1u);   // round-to-nearest-even
    return (unsigned short)(r >> 16);
}

// ---------------------------------------------------------------------------
// fp32 -> bf16 bulk convert
// ---------------------------------------------------------------------------
__global__ __launch_bounds__(256)
void cvt_f32_bf16(const float* __restrict__ in, unsigned short* __restrict__ out, int n) {
    int i = (blockIdx.x * blockDim.x + threadIdx.x) * 4;
    if (i + 3 < n) {
        float4 v = *(const float4*)&in[i];
        uint2 p;
        p.x = (unsigned)f2bf(v.x) | ((unsigned)f2bf(v.y) << 16);
        p.y = (unsigned)f2bf(v.z) | ((unsigned)f2bf(v.w) << 16);
        *(uint2*)&out[i] = p;
    }
}

// ---------------------------------------------------------------------------
// bf16 GEMM:  C[M,Nn] = A[M,K] * W[Nn,K]^T  (+bias)
// MODE 0: scatter bf16 into Q/K/V buffers laid out (B,H,N,D)
// MODE 1: write f32 to Out (final projection)
// Block 256 thr = 8 waves (2x4). Block tile 128x256, wave tile 64x64, BK=32.
// Double-buffered LDS, async global->LDS staging on CDNA5.
// ---------------------------------------------------------------------------
template<int MODE>
__global__ __launch_bounds__(256)
void gemm_bf16(const unsigned short* __restrict__ A,
               const unsigned short* __restrict__ W,
               const float* __restrict__ bias,
               unsigned short* __restrict__ Qb,
               unsigned short* __restrict__ Kb,
               unsigned short* __restrict__ Vb,
               float* __restrict__ Out,
               int M, int Nn, int Kd)
{
    constexpr int BM = 128, BN = 256, BK = 32;
    __shared__ __align__(16) unsigned short As[2][BM * BK];   // 2 x 8 KB
    __shared__ __align__(16) unsigned short Bs[2][BN * BK];   // 2 x 16 KB

    const int tid  = threadIdx.x;
    const int lane = tid & 31;
    const int wave = tid >> 5;          // 0..7
    const int wm   = wave >> 2;         // 0..1 -> 64 rows each
    const int wn   = wave & 3;          // 0..3 -> 64 cols each
    const int m0   = blockIdx.x * BM;
    const int n0   = blockIdx.y * BN;
    const int half = lane >> 4;         // 0 or 1
    const int l16  = lane & 15;
    const int koff = half * 8;          // ISA A/B bf16 lane->K mapping

    // stage one K-step: A tile 128x32 (512 16B chunks) + B tile 256x32 (1024)
    auto stage = [&](int kk, int buf) {
        #pragma unroll
        for (int i = 0; i < 6; ++i) {
            int c = tid + i * 256;                 // 0..1535, branch uniform per i
            if (c < 512) {
                int row = c >> 2, part = c & 3;
                cp16(&A[(size_t)(m0 + row) * Kd + kk + part * 8],
                     &As[buf][row * BK + part * 8]);
            } else {
                int cb = c - 512;
                int row = cb >> 2, part = cb & 3;
                cp16(&W[(size_t)(n0 + row) * Kd + kk + part * 8],
                     &Bs[buf][row * BK + part * 8]);
            }
        }
    };

    v8f acc[4][4] = {};
    int buf = 0;
    stage(0, 0);

    for (int kk = 0; kk < Kd; kk += BK) {
        wait_stage();
        __syncthreads();
        if (kk + BK < Kd) stage(kk + BK, buf ^ 1);

        // ---- fragments: 4 A + 4 B, then 16 WMMAs ----
        FragB a[4], b[4];
        #pragma unroll
        for (int mi = 0; mi < 4; ++mi) {
            int r = wm * 64 + mi * 16 + l16;
            a[mi].q[0] = *(const uint4*)&As[buf][r * BK + koff];
            a[mi].q[1] = *(const uint4*)&As[buf][r * BK + koff + 16];
        }
        #pragma unroll
        for (int ni = 0; ni < 4; ++ni) {
            int r = wn * 64 + ni * 16 + l16;
            b[ni].q[0] = *(const uint4*)&Bs[buf][r * BK + koff];
            b[ni].q[1] = *(const uint4*)&Bs[buf][r * BK + koff + 16];
        }
        #pragma unroll
        for (int mi = 0; mi < 4; ++mi)
            #pragma unroll
            for (int ni = 0; ni < 4; ++ni)
                acc[mi][ni] = __builtin_amdgcn_wmma_f32_16x16x32_bf16(
                    false, a[mi].v, false, b[ni].v, (short)0, acc[mi][ni], false, false);
        buf ^= 1;
    }

    // ---- epilogue ----
    #pragma unroll
    for (int mi = 0; mi < 4; ++mi) {
        #pragma unroll
        for (int ni = 0; ni < 4; ++ni) {
            int ncol = n0 + wn * 64 + ni * 16 + l16;
            float bv = bias[ncol];
            #pragma unroll
            for (int r = 0; r < 8; ++r) {
                int mrow = m0 + wm * 64 + mi * 16 + r + half * 8;
                float val = acc[mi][ni][r] + bv;
                if (MODE == 0) {
                    int s = ncol >> 10;            // 0:q 1:k 2:v
                    int h = (ncol >> 7) & 7;
                    int d = ncol & 127;
                    int bb = mrow >> 11;           // batch
                    int nn = mrow & 2047;          // seq pos
                    size_t addr = (((size_t)bb * NHEAD + h) * SEQ + nn) * HDIM + d;
                    unsigned short* dst = (s == 0) ? Qb : (s == 1) ? Kb : Vb;
                    dst[addr] = f2bf(val);
                } else {
                    Out[(size_t)mrow * Nn + ncol] = val;
                }
            }
        }
    }
}

// ---------------------------------------------------------------------------
// Banded flash attention.  Grid: (B*H, SEQ/64). Block: 128 thr = 4 waves,
// each wave owns 16 query rows. Key-chunk loop (32 keys) is uniform per block.
// ---------------------------------------------------------------------------
__global__ __launch_bounds__(128)
void attn_banded(const unsigned short* __restrict__ Q,
                 const unsigned short* __restrict__ K,
                 const unsigned short* __restrict__ V,
                 unsigned short* __restrict__ O)
{
    const int bh   = blockIdx.x;
    const int b    = bh >> 3, h = bh & 7;
    const int qb   = blockIdx.y * 64;
    const int tid  = threadIdx.x;
    const int lane = tid & 31;
    const int wave = tid >> 5;          // 0..3
    const int half = lane >> 4;
    const int l16  = lane & 15;
    const int koff = half * 8;
    const int q0   = qb + wave * 16;

    const size_t hoff = ((size_t)b * NHEAD + h) * SEQ * HDIM;
    const unsigned short* Qp = Q + hoff;
    const unsigned short* Kp = K + hoff;
    const unsigned short* Vp = V + hoff;

    __shared__ __align__(16) unsigned short Ks[32 * HDIM];     // 8 KB (key, d)
    __shared__ __align__(16) unsigned short Vt[HDIM * 32];     // 8 KB (d, key) transposed
    __shared__ __align__(16) unsigned short Ps[4][16 * 32];    // 4 KB P scratch per wave

    // Q fragments: 16 rows x 128 d -> 4 A-fragments, loaded once
    FragB qf[4];
    #pragma unroll
    for (int c = 0; c < 4; ++c) {
        size_t base = (size_t)(q0 + l16) * HDIM + c * 32 + koff;
        qf[c].q[0] = *(const uint4*)&Qp[base];
        qf[c].q[1] = *(const uint4*)&Qp[base + 16];
    }

    v8f  o[8] = {};
    float mrow[8], lrow[8];
    #pragma unroll
    for (int r = 0; r < 8; ++r) { mrow[r] = -1e30f; lrow[r] = 0.f; }

    int k0 = (qb >= WHALF) ? (qb - WHALF) : 0;
    int k1 = qb + 64 + WHALF; if (k1 > SEQ) k1 = SEQ;
    const float scale = 0.03125f;    // 1/sqrt(C) = 1/32

    for (int kc = k0; kc < k1; kc += 32) {
        __syncthreads();
        // ---- stage K (async, as-is) and V (sync, transposed): 32x128 each ----
        #pragma unroll
        for (int i = 0; i < 4; ++i) {
            int c    = tid + i * 128;          // 0..511
            int row  = c >> 4;                 // key 0..31
            int part = c & 15;                 // d-chunk of 8
            cp16(&Kp[(size_t)(kc + row) * HDIM + part * 8], &Ks[row * HDIM + part * 8]);
            uint4 vd = *(const uint4*)&Vp[(size_t)(kc + row) * HDIM + part * 8];
            const unsigned short* e = (const unsigned short*)&vd;
            #pragma unroll
            for (int j = 0; j < 8; ++j)
                Vt[(part * 8 + j) * 32 + row] = e[j];
        }
        wait_stage();
        __syncthreads();

        // ---- S = Q K^T over this 32-key chunk: two 16x16 tiles ----
        v8f s[2] = {};
        #pragma unroll
        for (int c = 0; c < 4; ++c) {          // contraction over d in 32-steps
            #pragma unroll
            for (int j = 0; j < 2; ++j) {
                FragB kf;
                int kr = j * 16 + l16;
                kf.q[0] = *(const uint4*)&Ks[kr * HDIM + c * 32 + koff];
                kf.q[1] = *(const uint4*)&Ks[kr * HDIM + c * 32 + koff + 16];
                s[j] = __builtin_amdgcn_wmma_f32_16x16x32_bf16(
                    false, qf[c].v, false, kf.v, (short)0, s[j], false, false);
            }
        }

        // ---- mask + scale + online softmax (row stats per lane-half) ----
        #pragma unroll
        for (int r = 0; r < 8; ++r) {
            int qrow = q0 + r + half * 8;
            float best = -1e30f;
            #pragma unroll
            for (int j = 0; j < 2; ++j) {
                int kcol = kc + j * 16 + l16;
                int dd   = qrow - kcol;
                float vv = (dd < WHALF && dd > -WHALF) ? s[j][r] * scale : -1e30f;
                s[j][r] = vv;
                best = fmaxf(best, vv);
            }
            #pragma unroll
            for (int off = 1; off < 16; off <<= 1)
                best = fmaxf(best, __shfl_xor(best, off, 32));
            float mn = fmaxf(mrow[r], best);
            float alpha = __expf(mrow[r] - mn);
            mrow[r] = mn;
            float psum = 0.f;
            #pragma unroll
            for (int j = 0; j < 2; ++j) {
                float p = __expf(s[j][r] - mn);
                s[j][r] = p;
                psum += p;
            }
            #pragma unroll
            for (int off = 1; off < 16; off <<= 1)
                psum += __shfl_xor(psum, off, 32);
            lrow[r] = lrow[r] * alpha + psum;
            #pragma unroll
            for (int t = 0; t < 8; ++t) o[t][r] *= alpha;
        }

        // ---- reshape P: C-layout -> A-fragment layout via per-wave LDS ----
        unsigned short* pw = Ps[wave];
        #pragma unroll
        for (int j = 0; j < 2; ++j)
            #pragma unroll
            for (int r = 0; r < 8; ++r)
                pw[(r + half * 8) * 32 + j * 16 + l16] = f2bf(s[j][r]);
        asm volatile("s_wait_dscnt 0" ::: "memory");   // wave-internal LDS RAW

        FragB pf;
        pf.q[0] = *(const uint4*)&pw[l16 * 32 + koff];
        pf.q[1] = *(const uint4*)&pw[l16 * 32 + koff + 16];

        // ---- O += P V : 8 d-tiles, V fragments from transposed LDS ----
        #pragma unroll
        for (int t = 0; t < 8; ++t) {
            FragB vf;
            int dr = t * 16 + l16;
            vf.q[0] = *(const uint4*)&Vt[dr * 32 + koff];
            vf.q[1] = *(const uint4*)&Vt[dr * 32 + koff + 16];
            o[t] = __builtin_amdgcn_wmma_f32_16x16x32_bf16(
                false, pf.v, false, vf.v, (short)0, o[t], false, false);
        }
    }

    // ---- finalize: O /= l, write bf16 into (B, N, H*D) for the projection ----
    #pragma unroll
    for (int r = 0; r < 8; ++r) {
        float inv = 1.0f / lrow[r];
        int qrow = q0 + r + half * 8;
        size_t rowbase = ((size_t)b * SEQ + qrow) * CH + h * HDIM;
        #pragma unroll
        for (int t = 0; t < 8; ++t)
            O[rowbase + t * 16 + l16] = f2bf(o[t][r] * inv);
    }
}

// ---------------------------------------------------------------------------
// Host-side orchestration
// ---------------------------------------------------------------------------
extern "C" void kernel_launch(void* const* d_in, const int* in_sizes, int n_in,
                              void* d_out, int out_size, void* d_ws, size_t ws_size,
                              hipStream_t stream) {
    const float* x      = (const float*)d_in[0];   // (B,N,C)
    const float* w_qkv  = (const float*)d_in[1];   // (3C,C)
    const float* b_qkv  = (const float*)d_in[2];   // (3C)
    const float* w_proj = (const float*)d_in[3];   // (C,C)
    const float* b_proj = (const float*)d_in[4];   // (C)
    float* out = (float*)d_out;

    const int nx  = MTOT * CH;        // 4,194,304
    const int nwq = 3 * CH * CH;      // 3,145,728
    const int nwp = CH * CH;          // 1,048,576
    const size_t nQKV = (size_t)BATCH * NHEAD * SEQ * HDIM;   // 4,194,304

    unsigned short* ws = (unsigned short*)d_ws;    // element (ushort) offsets below
    unsigned short* xb    = ws;
    unsigned short* wqkvb = xb    + nx;
    unsigned short* wprojb= wqkvb + nwq;
    unsigned short* Qb    = wprojb+ nwp;
    unsigned short* Kb    = Qb    + nQKV;
    unsigned short* Vb    = Kb    + nQKV;
    unsigned short* attb  = Vb    + nQKV;          // (B,N,C) bf16

    // 1) converts
    cvt_f32_bf16<<<nx  / 1024, 256, 0, stream>>>(x,      xb,     nx);
    cvt_f32_bf16<<<nwq / 1024, 256, 0, stream>>>(w_qkv,  wqkvb,  nwq);
    cvt_f32_bf16<<<nwp / 1024, 256, 0, stream>>>(w_proj, wprojb, nwp);

    // 2) QKV projection: M=4096, Nn=3072, K=1024
    gemm_bf16<0><<<dim3(MTOT / 128, (3 * CH) / 256), 256, 0, stream>>>(
        xb, wqkvb, b_qkv, Qb, Kb, Vb, nullptr, MTOT, 3 * CH, CH);

    // 3) banded attention
    attn_banded<<<dim3(BATCH * NHEAD, SEQ / 64), 128, 0, stream>>>(Qb, Kb, Vb, attb);

    // 4) output projection: M=4096, Nn=1024, K=1024 -> f32 + bias
    gemm_bf16<1><<<dim3(MTOT / 128, CH / 256), 256, 0, stream>>>(
        attb, wprojb, b_proj, nullptr, nullptr, nullptr, out, MTOT, CH, CH);

    (void)in_sizes; (void)n_in; (void)out_size; (void)ws_size;
}
